// TiledMemristorLinear_50826642981445
// MI455X (gfx1250) — 8-bit matrix operand order repaired and checked by reading, compiled, not run on hardware
//
#include <hip/hip_runtime.h>

typedef __attribute__((ext_vector_type(8))) int v8i;

#define IN_F   2048
#define OUT_F  2048
#define NPLANES 3
#define TILE_BYTES 65536

__global__ __launch_bounds__(256) void dac_quant_kernel(const float* __restrict__ x,
                                                        signed char* __restrict__ q,
                                                        int n4) {
    int i = blockIdx.x * blockDim.x + threadIdx.x;
    if (i >= n4) return;
    float4 v = ((const float4*)x)[i];
    char4 o;
    o.x = (signed char)(int)rintf(fminf(fmaxf(v.x, -1.f), 1.f) * 127.f);
    o.y = (signed char)(int)rintf(fminf(fmaxf(v.y, -1.f), 1.f) * 127.f);
    o.z = (signed char)(int)rintf(fminf(fmaxf(v.z, -1.f), 1.f) * 127.f);
    o.w = (signed char)(int)rintf(fminf(fmaxf(v.w, -1.f), 1.f) * 127.f);
    ((char4*)q)[i] = o;
}

__global__ __launch_bounds__(256) void wpack_kernel(const float* __restrict__ W,
                                                    signed char* __restrict__ wt) {
    const int tile = blockIdx.x;
    const int n    = threadIdx.x;
    const float* src = W + (size_t)tile * TILE_BYTES;
    signed char* dst = wt + (size_t)tile * TILE_BYTES + (size_t)n * 256;
    for (int mb = 0; mb < 64; ++mb) {
        char4 o;
        o.x = (signed char)(int)rintf(src[(mb * 4 + 0) * 256 + n]);
        o.y = (signed char)(int)rintf(src[(mb * 4 + 1) * 256 + n]);
        o.z = (signed char)(int)rintf(src[(mb * 4 + 2) * 256 + n]);
        o.w = (signed char)(int)rintf(src[(mb * 4 + 3) * 256 + n]);
        *(char4*)(dst + mb * 4) = o;
    }
}

__global__ __launch_bounds__(256) void memristor_wmma_kernel(
    const signed char* __restrict__ q,
    const signed char* __restrict__ wt,
    const float* __restrict__ bias,
    float* __restrict__ out,
    int rows)
{
    const int lane = threadIdx.x & 31;
    const int wave = threadIdx.x >> 5;
    const int hi   = lane >> 4;
    const int l16  = lane & 15;

    const int row0  = blockIdx.y * 32 + (wave & 1) * 16;
    const int col0  = blockIdx.x * 256 + (wave >> 1) * 64;
    const int ktile = col0 >> 8;
    const int nIn   = col0 & 255;

    const signed char* aRow = q + (size_t)(row0 + l16) * IN_F;

    float accf[4][8];
    #pragma unroll
    for (int t = 0; t < 4; ++t)
        #pragma unroll
        for (int r = 0; r < 8; ++r) accf[t][r] = 0.f;

    int bOff[4];
    #pragma unroll
    for (int t = 0; t < 4; ++t) bOff[t] = (nIn + t * 16 + l16) * 256 + hi * 16;

    for (int j = 0; j < 8; ++j) {
        const signed char* aBase = aRow + j * 256 + hi * 16;
        union { v8i v; int2 d[4]; } A[4];
        #pragma unroll
        for (int ks = 0; ks < 4; ++ks) {
            const signed char* a = aBase + ks * 64;
            A[ks].d[0] = *(const int2*)(a);
            A[ks].d[1] = *(const int2*)(a + 8);
            A[ks].d[2] = *(const int2*)(a + 32);
            A[ks].d[3] = *(const int2*)(a + 40);
        }

        #pragma unroll
        for (int l = 0; l < NPLANES; ++l) {
            const float post = (float)(1 << (2 - l)) * (1.0f / 256.0f);
            const signed char* bTile = wt + (size_t)((l * 8 + j) * 8 + ktile) * TILE_BYTES;

            union { v8i v; int i[8]; } acc[4];
            #pragma unroll
            for (int t = 0; t < 4; ++t)
                #pragma unroll
                for (int r = 0; r < 8; ++r) acc[t].i[r] = 0;

            #pragma unroll
            for (int ks = 0; ks < 4; ++ks) {
                #pragma unroll
                for (int t = 0; t < 4; ++t) {
                    union { v8i v; int4 d[2]; } B;
                    const signed char* b = bTile + bOff[t] + ks * 64;
                    B.d[0] = *(const int4*)(b);
                    B.d[1] = *(const int4*)(b + 32);
                    acc[t].v = __builtin_amdgcn_wmma_i32_16x16x64_iu8(
                        true, A[ks].v, true, B.v, acc[t].v, false, false);
                }
            }

            #pragma unroll
            for (int t = 0; t < 4; ++t)
                #pragma unroll
                for (int r = 0; r < 8; ++r) {
                    float y = (float)acc[t].i[r] * (256.0f / 127.0f);
                    y = rintf(y);
                    y = fminf(fmaxf(y, -4096.0f), 4096.0f);
                    accf[t][r] = fmaf(y, post, accf[t][r]);
                }
        }
    }

    #pragma unroll
    for (int t = 0; t < 4; ++t) {
        const int col = col0 + t * 16 + l16;
        const float bs = bias[col];
        #pragma unroll
        for (int r = 0; r < 8; ++r) {
            const int row = row0 + hi * 8 + r;
            out[(size_t)row * OUT_F + col] = fmaf(accf[t][r], 0.01f, bs);
        }
    }
}

extern "C" void kernel_launch(void* const* d_in, const int* in_sizes, int n_in,
                              void* d_out, int out_size, void* d_ws, size_t ws_size,
                              hipStream_t stream) {
    const float* x    = (const float*)d_in[0];
    const float* W    = (const float*)d_in[1];
    const float* bias = (const float*)d_in[2];
    float* out = (float*)d_out;

    const int rows = in_sizes[0] / IN_F;

    signed char* q  = (signed char*)d_ws;
    signed char* wt = (signed char*)d_ws + (size_t)rows * IN_F;

    const int n4 = in_sizes[0] / 4;
    dac_quant_kernel<<<(n4 + 255) / 256, 256, 0, stream>>>(x, q, n4);
    wpack_kernel<<<192, 256, 0, stream>>>(W, wt);

    dim3 grid(OUT_F / 256, rows / 32);
    memristor_wmma_kernel<<<grid, 256, 0, stream>>>(q, wt, bias, out, rows);
}
